// Network_21337397527096
// MI455X (gfx1250) — matrix operand order repaired and checked by reading, not compiled, not run on hardware
//
#include <hip/hip_runtime.h>

typedef __attribute__((ext_vector_type(16))) _Float16 v16h;
typedef __attribute__((ext_vector_type(8)))  _Float16 v8h;
typedef __attribute__((ext_vector_type(8)))  float    v8f;

#define SNN_THRESHOLD 1.25f
#define T_TOTAL 2048
#define TT 128  // time tile: 8 N-tiles of 16; each of 4 waves owns 2

// A-fragment loader, this code's contiguous k order (B at line 44 uses the same):
// lanes 0-15 hold K = [0..15], lanes 16-31 hold K = [16..31], contiguous.
__device__ __forceinline__ v16h load_a_frag(const _Float16* w, int Kpad,
                                            int m, int kk, int half) {
  const v8h lo = *(const v8h*)(w + m * Kpad + kk * 32 + half * 16);
  const v8h hi = *(const v8h*)(w + m * Kpad + kk * 32 + half * 16 + 8);
  return __builtin_shufflevector(lo, hi, 0, 1, 2, 3, 4, 5, 6, 7,
                                 8, 9, 10, 11, 12, 13, 14, 15);
}

// Fused GEMM stage over one TT-step tile: z[t][c] = sum_k W[c][k] * S[t][k].
// in : LDS activation buffer [TT][Kpad] f16, t-major
// w  : LDS weights [M][Kpad] f16 (zero padded)
// zb : LDS pre-activation buffer [TT][64] f32
template <int M, int Kpad>
__device__ __forceinline__ void gemm_tile(const _Float16* w, const _Float16* in,
                                          float* zb, int lane, int wave) {
  const int row  = lane & 15;   // M index (A) / N(time) index (B)
  const int half = lane >> 4;
#pragma unroll
  for (int mt = 0; mt < M / 16; ++mt) {
    v16h a[Kpad / 32];
#pragma unroll
    for (int kk = 0; kk < Kpad / 32; ++kk)
      a[kk] = load_a_frag(w, Kpad, mt * 16 + row, kk, half);
#pragma unroll
    for (int nt = 0; nt < TT / 64; ++nt) {        // N-tiles owned by this wave
      const int n0 = (wave + nt * 4) * 16;
      v8f acc = {};
#pragma unroll
      for (int kk = 0; kk < Kpad / 32; ++kk) {
        // B in the same contiguous k order as A: lane n -> column n; lanes 0-15 K 0..15,
        // lanes 16-31 K 16..31, contiguous.
        v16h b = *(const v16h*)(in + (n0 + row) * Kpad + kk * 32 + half * 16);
        acc = __builtin_amdgcn_wmma_f32_16x16x32_f16(
            /*neg_a=*/false, a[kk], /*neg_b=*/false, b,
            /*c_mod=*/(short)0, acc, /*reuse_a=*/false, /*reuse_b=*/false);
      }
      // D layout: VGPR r -> (M = mt*16 + 8*half + r, N = n0 + (lane&15))
      const int nidx  = n0 + row;
      const int mbase = mt * 16 + half * 8;
#pragma unroll
      for (int r = 0; r < 8; ++r)
        zb[nidx * 64 + mbase + r] = acc[r];
    }
  }
}

__global__ void __launch_bounds__(128, 1) snn_fused_kernel(
    const float* __restrict__ x,                                   // [B,11,T]
    const float* __restrict__ v0, const float* __restrict__ g0, const float* __restrict__ d0,
    const float* __restrict__ v1, const float* __restrict__ g1, const float* __restrict__ d1,
    const float* __restrict__ v2, const float* __restrict__ g2, const float* __restrict__ d2,
    const float* __restrict__ v3, const float* __restrict__ g3,
    const float* __restrict__ cd, const float* __restrict__ vd,
    float* __restrict__ out)                                       // [B,5,T]
{
  __shared__ __align__(32) _Float16 w0[32 * 32];   // 2 KB  (K pad 11->32)
  __shared__ __align__(32) _Float16 w1[64 * 32];   // 4 KB
  __shared__ __align__(32) _Float16 w2[32 * 64];   // 4 KB
  __shared__ __align__(32) _Float16 w3[16 * 32];   // 1 KB  (5->16 rows, zero pad)
  __shared__ __align__(32) _Float16 sIn[TT * 32];  // 8 KB  layer0 input (pad zeroed once)
  __shared__ __align__(32) _Float16 sA[TT * 64];   // 16 KB layer1 spikes (64 ch)
  __shared__ __align__(32) _Float16 sB[TT * 32];   // 8 KB  layer0/layer2 spikes (32 ch)
  __shared__ float zbuf[TT * 64];                  // 32 KB pre-activations
  __shared__ float sOut[5 * TT];                   // 2.5 KB output staging

  const int b    = blockIdx.x;
  const int tid  = threadIdx.x;
  const int lane = tid & 31;
  const int wave = tid >> 5;

  // ---- weight-norm prep: w = g * v / ||v||, f16, zero-padded, into LDS ----
  for (int r = tid; r < 144; r += 128) {
    const float* vv; const float* gg; _Float16* wdst;
    int Cin, Kpad, o, Mreal;
    if (r < 32)       { vv = v0; gg = g0; wdst = w0; Cin = 11; Kpad = 32; o = r;       Mreal = 32; }
    else if (r < 96)  { vv = v1; gg = g1; wdst = w1; Cin = 32; Kpad = 32; o = r - 32;  Mreal = 64; }
    else if (r < 128) { vv = v2; gg = g2; wdst = w2; Cin = 64; Kpad = 64; o = r - 96;  Mreal = 32; }
    else              { vv = v3; gg = g3; wdst = w3; Cin = 32; Kpad = 32; o = r - 128; Mreal = 5;  }
    if (o < Mreal) {
      float ss = 0.f;
      for (int c = 0; c < Cin; ++c) { float t = vv[o * Cin + c]; ss += t * t; }
      const float scale = gg[o] * rsqrtf(ss);
      for (int c = 0; c < Kpad; ++c)
        wdst[o * Kpad + c] = (_Float16)((c < Cin) ? vv[o * Cin + c] * scale : 0.f);
    } else {
      for (int c = 0; c < Kpad; ++c) wdst[o * Kpad + c] = (_Float16)0.f;
    }
  }
  // zero layer0 input buffer once (pad channels 11..31 stay zero forever)
  for (int i = tid; i < TT * 32; i += 128) sIn[i] = (_Float16)0.f;

  // ---- per-thread LIF parameters and persistent state ----
  const float omc0 = 1.f - cd[0], omv0 = 1.f - vd[0];
  const float omc1 = 1.f - cd[1], omv1 = 1.f - vd[1];
  const float omc2 = 1.f - cd[2], omv2 = 1.f - vd[2];
  const float omc3 = 1.f - cd[3], omv3 = 1.f - vd[3];
  const bool sh0 = (tid < 32) && (d0[tid] > 0.5f);
  const bool sh1 = (tid < 64) && (d1[tid] > 0.5f);
  const bool sh2 = (tid < 32) && (d2[tid] > 0.5f);

  float cur0 = 0.f, vol0 = 0.f, prev0 = 0.f;
  float cur1 = 0.f, vol1 = 0.f, prev1 = 0.f;
  float cur2 = 0.f, vol2 = 0.f, prev2 = 0.f;
  float cur3 = 0.f, vol3 = 0.f;

  __syncthreads();

  for (int t0 = 0; t0 < T_TOTAL; t0 += TT) {
    // ---- coalesced input tile load: lanes walk consecutive t ----
    for (int i = tid; i < 11 * TT; i += 128) {
      const int c = i / TT, tt = i % TT;
      sIn[tt * 32 + c] = (_Float16)x[((long)b * 11 + c) * T_TOTAL + t0 + tt];
    }
    __syncthreads();

    // ================= layer 0: 11->32 =================
    gemm_tile<32, 32>(w0, sIn, zbuf, lane, wave);
    __syncthreads();
    if (tid < 32) {
      float sp = prev0;
#pragma unroll 4
      for (int t = 0; t < TT; ++t) {
        cur0 = cur0 * omc0 + zbuf[t * 64 + tid];
        vol0 = vol0 * omv0 + cur0;
        const float s = (vol0 >= SNN_THRESHOLD) ? 1.f : 0.f;
        vol0 -= vol0 * s;
        sB[t * 32 + tid] = (_Float16)(sh0 ? sp : s);
        sp = s;
      }
      prev0 = sp;
    }
    __syncthreads();

    // ================= layer 1: 32->64 =================
    gemm_tile<64, 32>(w1, sB, zbuf, lane, wave);
    __syncthreads();
    if (tid < 64) {
      float sp = prev1;
#pragma unroll 4
      for (int t = 0; t < TT; ++t) {
        cur1 = cur1 * omc1 + zbuf[t * 64 + tid];
        vol1 = vol1 * omv1 + cur1;
        const float s = (vol1 >= SNN_THRESHOLD) ? 1.f : 0.f;
        vol1 -= vol1 * s;
        sA[t * 64 + tid] = (_Float16)(sh1 ? sp : s);
        sp = s;
      }
      prev1 = sp;
    }
    __syncthreads();

    // ================= layer 2: 64->32 =================
    gemm_tile<32, 64>(w2, sA, zbuf, lane, wave);
    __syncthreads();
    if (tid < 32) {
      float sp = prev2;
#pragma unroll 4
      for (int t = 0; t < TT; ++t) {
        cur2 = cur2 * omc2 + zbuf[t * 64 + tid];
        vol2 = vol2 * omv2 + cur2;
        const float s = (vol2 >= SNN_THRESHOLD) ? 1.f : 0.f;
        vol2 -= vol2 * s;
        sB[t * 32 + tid] = (_Float16)(sh2 ? sp : s);
        sp = s;
      }
      prev2 = sp;
    }
    __syncthreads();

    // ================= layer 3: 32->5 (no delay) =================
    gemm_tile<16, 32>(w3, sB, zbuf, lane, wave);
    __syncthreads();
    if (tid < 5) {
#pragma unroll 4
      for (int t = 0; t < TT; ++t) {
        cur3 = cur3 * omc3 + zbuf[t * 64 + tid];
        vol3 = vol3 * omv3 + cur3;
        const float s = (vol3 >= SNN_THRESHOLD) ? 1.f : 0.f;
        vol3 -= vol3 * s;
        sOut[tid * TT + t] = s;
      }
    }
    __syncthreads();
    // ---- coalesced output store: lanes walk consecutive t ----
    for (int i = tid; i < 5 * TT; i += 128) {
      const int c = i / TT, tt = i % TT;
      out[((long)b * 5 + c) * T_TOTAL + t0 + tt] = sOut[i];
    }
    __syncthreads();
  }
}

extern "C" void kernel_launch(void* const* d_in, const int* in_sizes, int n_in,
                              void* d_out, int out_size, void* d_ws, size_t ws_size,
                              hipStream_t stream) {
  const float* x  = (const float*)d_in[0];
  const float* v0 = (const float*)d_in[1];
  const float* g0 = (const float*)d_in[2];
  const float* d0 = (const float*)d_in[3];
  const float* v1 = (const float*)d_in[4];
  const float* g1 = (const float*)d_in[5];
  const float* d1 = (const float*)d_in[6];
  const float* v2 = (const float*)d_in[7];
  const float* g2 = (const float*)d_in[8];
  const float* d2 = (const float*)d_in[9];
  const float* v3 = (const float*)d_in[10];
  const float* g3 = (const float*)d_in[11];
  const float* cd = (const float*)d_in[12];
  const float* vd = (const float*)d_in[13];
  float* out = (float*)d_out;

  const int B = in_sizes[0] / (11 * T_TOTAL);  // 512
  snn_fused_kernel<<<dim3(B), dim3(128), 0, stream>>>(
      x, v0, g0, d0, v1, g1, d1, v2, g2, d2, v3, g3, cd, vd, out);
}
